// JointHistLayer_56092272885906
// MI455X (gfx1250) — matrix operand order repaired and checked by reading, not compiled, not run on hardware
//
#include <hip/hip_runtime.h>
#include <stdint.h>

typedef __attribute__((ext_vector_type(16))) __bf16 v16bf;
typedef __attribute__((ext_vector_type(8)))  float  v8f;
typedef __attribute__((ext_vector_type(4)))  unsigned v4u;
typedef __attribute__((ext_vector_type(8)))  int      v8i;
typedef __attribute__((ext_vector_type(4)))  int      v4i;

#define KBINS 256
#define NPIX  65536
#define CHUNK 32
#define SLAB  1024                      // pixels per TDM slab
#define CHPS  (SLAB / CHUNK)            // chunks per slab
#define BKK   (8 * KBINS * KBINS)       // full output element count

#if defined(__has_builtin)
#if __has_builtin(__builtin_amdgcn_tensor_load_to_lds) && \
    __has_builtin(__builtin_amdgcn_s_wait_tensorcnt)
#define HAVE_TDM 1
#endif
#endif

// pack two floats into one dword of two bf16 (round-to-nearest-even), pure VALU
__device__ __forceinline__ unsigned pack_bf16(float lo, float hi) {
    union { float f; unsigned u; } a, b;
    a.f = lo; b.f = hi;
    unsigned l = (a.u + 0x7FFFu + ((a.u >> 16) & 1u)) >> 16;
    unsigned h = (b.u + 0x7FFFu + ((b.u >> 16) & 1u)) >> 16;
    return (h << 16) | (l & 0xFFFFu);
}

__device__ __forceinline__ float sigf(float z) {
    return __builtin_amdgcn_rcpf(1.0f + __expf(-z));   // v_exp_f32 + v_rcp_f32
}

#ifdef HAVE_TDM
// TDM: DMA a 1-D slab of SLAB f32 from global into LDS. D# per ISA 8.3/8.4:
// group0: count=1 | lds_addr | global_addr | type=2
// group1: data_size=4B, tensor_dim0=tile_dim0=SLAB, dim1=1, stride0=SLAB
__device__ __forceinline__ void tdm_load_slab(unsigned lds_addr, const float* gsrc) {
    unsigned long long ga = (unsigned long long)(uintptr_t)gsrc;
    v4u g0 = { 1u,                                   // count=1, user mode
               lds_addr,                             // lds_addr[31:0]
               (unsigned)ga,                         // global_addr[31:0]
               (unsigned)((ga >> 32) & 0x1FFFFFFu) | (2u << 30) };  // ga[56:32], type=2
    v8i g1 = { 0x00020000,                           // data_size=2 (4B), mask=0
               (int)((SLAB & 0xFFFF) << 16),         // tensor_dim0[15:0]
               0x00010000,                           // tensor_dim0[31:16]=0, tensor_dim1=1
               (int)((SLAB & 0xFFFF) << 16),         // tile_dim0=SLAB
               0x00000001,                           // tile_dim1=1, tile_dim2=0
               (int)SLAB,                            // tensor_dim0_stride[31:0]
               (int)((SLAB & 0xFFFF) << 16),         // stride0 hi=0, stride1[15:0]=SLAB
               0 };
    v4i gz = { 0, 0, 0, 0 };
#if __clang_major__ >= 23
    v8i gz8 = { 0, 0, 0, 0, 0, 0, 0, 0 };
    __builtin_amdgcn_tensor_load_to_lds(g0, g1, gz, gz, gz8, 0);
#else
    __builtin_amdgcn_tensor_load_to_lds(g0, g1, gz, gz, 0);
#endif
}
#endif

__device__ __forceinline__ unsigned lds_offset(const void* p) {
    return (unsigned)(uintptr_t)p;   // LDS aperture: addr[31:0] = LDS offset
}

__global__ __launch_bounds__(256) void joint_hist_wmma(
        const float* __restrict__ x, const float* __restrict__ y,
        float* __restrict__ dst, int nspan, float scale, size_t partStride) {
    __shared__ float xslab[2][SLAB];                   // double-buffered pixel slabs
    __shared__ float yslab[2][SLAB];
    __shared__ __align__(32) unsigned aimg[4 * 256];   // 4 k-blocks  x (32 lanes x 16 bf16)
    __shared__ __align__(32) unsigned bimg[16 * 256];  // 16 j-blocks x (32 lanes x 16 bf16)

    const int t    = threadIdx.x;
    const int lane = t & 31;
    const int wave = t >> 5;
    const int kb   = wave & 3;             // this wave's k-block within slab
    const int jb0  = (wave >> 2) * 8;      // first of 8 j-blocks for this wave
    const int kbase = blockIdx.x * 64;     // k-slab base (4 cover K=256)
    const int b     = blockIdx.y;          // batch
    const int nbeg  = blockIdx.z * nspan;  // pixel split start
    const int nslab = nspan / SLAB;

    const float* xb = x + (size_t)b * NPIX;
    const float* yb = y + (size_t)b * NPIX;
    float* outp = dst + (size_t)blockIdx.z * partStride;

    v8f acc[8];
    const v8f vzero = {0.f, 0.f, 0.f, 0.f, 0.f, 0.f, 0.f, 0.f};
#pragma unroll
    for (int i = 0; i < 8; ++i) acc[i] = vzero;

    // ---- image-builder decode: thread owns n-pair p (n0=2p,n0+1) and a run of
    // adjacent bins (4 k-rows for A, 16 j-rows for B) so sigmoids telescope.
    const int p   = t & 15;
    const int g   = t >> 4;
    const int n0i = 2 * p;

    // B-matrix 32x16 bf16 layout: lane = (j&15) + ((n&8) ? 16:0); elem e = (n&7) | ((n&16)>>1)
    const unsigned b_hsel  = (n0i & 8) ? 16u : 0u;
    const unsigned b_epair = (unsigned)(((n0i & 7) | ((n0i & 16) >> 1)) >> 1);
    const unsigned bbase   = (unsigned)g * 256u + b_hsel * 8u + b_epair;

    // A-matrix 16x32 bf16 layout (ISA 7.12.2)
    const unsigned a_e     = (unsigned)((n0i & 7) | ((n0i & 16) >> 1));
    const unsigned a_hsel  = (n0i & 8) ? 16u : 0u;
    const unsigned a_epair = a_e >> 1;
    const unsigned akb     = (unsigned)(g >> 2);
    const unsigned aklo    = (unsigned)((g & 3) * 4);
    const unsigned abase   = akb * 256u + (aklo + a_hsel) * 8u + a_epair;

    const v16bf* afrag = reinterpret_cast<const v16bf*>(aimg);
    const v16bf* bfrag = reinterpret_cast<const v16bf*>(bimg);

#ifdef HAVE_TDM
    if (wave == 0) {   // prologue: DMA slab 0 into buffer 0 (TENSORcnt +2)
        tdm_load_slab(lds_offset(&xslab[0][0]), xb + nbeg);
        tdm_load_slab(lds_offset(&yslab[0][0]), yb + nbeg);
    }
#endif

#pragma unroll 1
    for (int s = 0; s < nslab; ++s) {
        const int cur = s & 1;
        __syncthreads();   // all waves done reading buffer cur^1 (re-target of DMA)
#ifdef HAVE_TDM
        if (wave == 0) {
            if (s + 1 < nslab) {
                tdm_load_slab(lds_offset(&xslab[cur ^ 1][0]), xb + nbeg + (s + 1) * SLAB);
                tdm_load_slab(lds_offset(&yslab[cur ^ 1][0]), yb + nbeg + (s + 1) * SLAB);
                __builtin_amdgcn_s_wait_tensorcnt((short)2);  // slab s done (in-order)
            } else {
                __builtin_amdgcn_s_wait_tensorcnt((short)0);
            }
        }
#else
        for (int i = t; i < SLAB; i += 256) {   // fallback: cooperative copy
            xslab[cur][i] = xb[nbeg + s * SLAB + i];
            yslab[cur][i] = yb[nbeg + s * SLAB + i];
        }
#endif
        __syncthreads();   // slab cur visible to all waves

#pragma unroll 1
        for (int c = 0; c < CHPS; ++c) {
            if (c) __syncthreads();   // previous chunk's fragment reads complete
            const float* xs = &xslab[cur][c * CHUNK];
            const float* ys = &yslab[cur][c * CHUNK];

            // ---- A image: k rows kbase+4g..+3, two pixels; 2x5 sigmoids
            {
                const float xv0 = xs[n0i], xv1 = xs[n0i + 1];
                const float k0  = (float)(kbase + 4 * g);
                const float base0 = fmaf(xv0, 640.0f, -2.5f * k0);
                const float base1 = fmaf(xv1, 640.0f, -2.5f * k0);
                float sp0 = sigf(base0), sp1 = sigf(base1);
#pragma unroll
                for (int i = 0; i < 4; ++i) {
                    float sn0 = sigf(base0 - 2.5f * (float)(i + 1));
                    float sn1 = sigf(base1 - 2.5f * (float)(i + 1));
                    aimg[abase + (unsigned)i * 8u] = pack_bf16(sp0 - sn0, sp1 - sn1);
                    sp0 = sn0; sp1 = sn1;
                }
            }
            // ---- B image: j rows 16g..16g+15, two pixels; 2x17 sigmoids
            {
                const float yv0 = ys[n0i], yv1 = ys[n0i + 1];
                const float j0  = (float)(16 * g);
                const float base0 = fmaf(yv0, 640.0f, -2.5f * j0);
                const float base1 = fmaf(yv1, 640.0f, -2.5f * j0);
                float sp0 = sigf(base0), sp1 = sigf(base1);
#pragma unroll
                for (int i = 0; i < 16; ++i) {
                    float sn0 = sigf(base0 - 2.5f * (float)(i + 1));
                    float sn1 = sigf(base1 - 2.5f * (float)(i + 1));
                    bimg[bbase + (unsigned)i * 8u] = pack_bf16(sp0 - sn0, sp1 - sn1);
                    sp0 = sn0; sp1 = sn1;
                }
            }
            __syncthreads();

            // Each wave: one A fragment (2x ds_load_b128), 8 WMMAs vs 8 B fragments
            v16bf a = afrag[kb * 32 + lane];
#pragma unroll
            for (int tt = 0; tt < 8; ++tt) {
                v16bf bm = bfrag[(jb0 + tt) * 32 + lane];
                acc[tt] = __builtin_amdgcn_wmma_f32_16x16x32_bf16(
                    false, a, false, bm, (short)0, acc[tt], false, false);
            }
        }
    }

    // Epilogue: D layout — VGPR r: rows r (lanes 0-15) / r+8 (lanes 16-31)
#pragma unroll
    for (int tt = 0; tt < 8; ++tt) {
        const int jcol = (jb0 + tt) * 16 + (lane & 15);
        const int kr0  = kbase + kb * 16 + ((lane >> 4) << 3);
        float* o = outp + ((size_t)b * KBINS + kr0) * KBINS + jcol;
#pragma unroll
        for (int r = 0; r < 8; ++r)
            o[(size_t)r * KBINS] = acc[tt][r] * scale;
    }
}

// Deterministic reduction of S partial histograms (fixed summation order)
__global__ __launch_bounds__(256) void reduce_parts(
        const float* __restrict__ ws, float* __restrict__ out, int S) {
    const int idx = blockIdx.x * 256 + threadIdx.x;
    float s = 0.0f;
    for (int i = 0; i < S; ++i) s += ws[(size_t)i * BKK + idx];
    out[idx] = s * (1.0f / (float)NPIX);
}

extern "C" void kernel_launch(void* const* d_in, const int* in_sizes, int n_in,
                              void* d_out, int out_size, void* d_ws, size_t ws_size,
                              hipStream_t stream) {
    (void)in_sizes; (void)n_in; (void)out_size;
    const float* x = (const float*)d_in[0];   // (8,256,256) f32
    const float* y = (const float*)d_in[1];   // (8,256,256) f32
    float* out = (float*)d_out;               // (8,256,256) f32

    const size_t partBytes = (size_t)BKK * sizeof(float);  // 2 MB per split
    int S = 1;
    if      (ws_size >= 16 * partBytes) S = 16;
    else if (ws_size >=  8 * partBytes) S = 8;
    else if (ws_size >=  4 * partBytes) S = 4;
    else if (ws_size >=  2 * partBytes) S = 2;

    dim3 block(256, 1, 1);                    // 8 wave32 waves
    if (S > 1) {
        float* ws = (float*)d_ws;
        dim3 grid(4, 8, S);                   // k-slabs x batches x pixel-splits
        joint_hist_wmma<<<grid, block, 0, stream>>>(
            x, y, ws, NPIX / S, 1.0f, (size_t)BKK);
        reduce_parts<<<dim3(BKK / 256, 1, 1), block, 0, stream>>>(ws, out, S);
    } else {
        dim3 grid(4, 8, 1);
        joint_hist_wmma<<<grid, block, 0, stream>>>(
            x, y, out, NPIX, 1.0f / (float)NPIX, 0);
    }
}
